// my_model3_35012573397310
// MI455X (gfx1250) — matrix operand order repaired and checked by reading, not compiled, not run on hardware
//
#include <hip/hip_runtime.h>
#include <hip/hip_bf16.h>
#include <cstdint>
#include <cstddef>

// ---------------- problem constants (match reference) ----------------
#define N_NODES 65536
#define N_EDGES 32768
#define D_CTX   768
#define D_REL   97
#define D_PAIR  194          // 2*D_REL
#define NT      7            // N tiles: 7*16 = 112 >= 97
#define REL_STRIDE 208       // padded 194 -> 208 halves (26 * 8)
#define PROJ_STRIDE 112
#define KC_PROJ 24           // 768 / 32
#define JC_GATE 7            // ceil(194/32)
#define KSTEPS_GATE (D_PAIR * JC_GATE)   // 1358
#define SD_STRIDE 232        // LDS row stride (halves), conflict-friendly

typedef __attribute__((ext_vector_type(16))) _Float16 v16h;
typedef __attribute__((ext_vector_type(8)))  _Float16 v8h;
typedef __attribute__((ext_vector_type(2)))  _Float16 h2;
typedef __attribute__((ext_vector_type(8)))  float    v8f;

union A16 { v16h v; h2 p[8]; };
union B8  { v8h  v; h2 p[4]; };
union Pack16 { _Float16 h[16]; int4 q[2]; };

// ============================================================================
// Pack W_bl[0] (768x97 f32) into WMMA B-fragment-native f16 layout.
// frag f = kc*NT + nt ; per-lane 32B chunk holds 16 halves in VGPR order.
// B k order used here (same as A): lane n = L&15 ; lanes 0-15 -> K 0..15, lanes 16-31 -> K 16..31;
// reg pair v -> K = kb2 + 2v, kb2+2v+1.
// ============================================================================
__global__ void pack_projB_kernel(const float* __restrict__ Wbl,
                                  _Float16* __restrict__ outp) {
    int idx = blockIdx.x * blockDim.x + threadIdx.x;
    const int total = KC_PROJ * NT * 32;
    if (idx >= total) return;
    int frag = idx >> 5, lane = idx & 31;
    int kc = frag / NT, nt = frag % NT;
    int nn = lane & 15;
    int kb2 = (lane & 16) ? 16 : 0;
    int col = nt * 16 + nn;
    Pack16 buf;
#pragma unroll
    for (int q = 0; q < 8; ++q) {
        int k0 = kc * 32 + kb2 + 2 * q;
        float a = (col < D_REL) ? Wbl[(size_t)k0 * D_REL + col] : 0.0f;
        float b = (col < D_REL) ? Wbl[(size_t)(k0 + 1) * D_REL + col] : 0.0f;
        buf.h[2 * q]     = (_Float16)a;
        buf.h[2 * q + 1] = (_Float16)b;
    }
    int4* dst = (int4*)(outp + (size_t)frag * 512 + lane * 16);
    dst[0] = buf.q[0];
    dst[1] = buf.q[1];
}

// ============================================================================
// Pack W_bl1 (97x194x194 f32) into fragment-native f16 layout for the big
// implicit GEMM: contraction index ij ordered i-major, j chunked by 32.
// kstep = i*7 + jc ; frag = kstep*NT + nt ; B[kk][n] = W[nt*16+n][i][jc*32+kk].
// ============================================================================
__global__ void pack_Wt_kernel(const float* __restrict__ W1,
                               _Float16* __restrict__ outp) {
    int idx = blockIdx.x * blockDim.x + threadIdx.x;
    const int total = KSTEPS_GATE * NT * 32;
    if (idx >= total) return;
    int frag = idx >> 5, lane = idx & 31;
    int kstep = frag / NT, nt = frag % NT;
    int i = kstep / JC_GATE, jc = kstep % JC_GATE;
    int nn = lane & 15;
    int kb2 = (lane & 16) ? 16 : 0;
    int col = nt * 16 + nn;                 // output rel dim k (0..96 valid)
    Pack16 buf;
#pragma unroll
    for (int q = 0; q < 8; ++q) {
        int j0 = jc * 32 + kb2 + 2 * q;
        int j1 = j0 + 1;
        float a = (col < D_REL && j0 < D_PAIR)
                      ? W1[((size_t)col * D_PAIR + i) * D_PAIR + j0] : 0.0f;
        float b = (col < D_REL && j1 < D_PAIR)
                      ? W1[((size_t)col * D_PAIR + i) * D_PAIR + j1] : 0.0f;
        buf.h[2 * q]     = (_Float16)a;
        buf.h[2 * q + 1] = (_Float16)b;
    }
    int4* dst = (int4*)(outp + (size_t)frag * 512 + lane * 16);
    dst[0] = buf.q[0];
    dst[1] = buf.q[1];
}

// ============================================================================
// proj = n @ W_bl[0] : WMMA f16 GEMM, M=65536, N=97(pad 112), K=768.
// 2 waves / block, each wave does a 16-row strip (all 7 N tiles).
// ============================================================================
__global__ void __launch_bounds__(64)
proj_kernel(const float* __restrict__ nmat,
            const _Float16* __restrict__ projB,
            float* __restrict__ proj) {
    __shared__ __align__(16) _Float16 nsm[2][16 * D_CTX];
    int wave = threadIdx.x >> 5, lane = threadIdx.x & 31;
    size_t rowBase = (size_t)blockIdx.x * 32 + (size_t)wave * 16;

    // stage 16 rows of n as f16 (coalesced float4 loads)
    for (int r = 0; r < 16; ++r) {
        const float* np = nmat + (rowBase + r) * D_CTX;
        _Float16* dp = &nsm[wave][r * D_CTX];
        for (int t = 0; t < 6; ++t) {
            int c4 = (t * 32 + lane) * 4;
            float4 f = *(const float4*)(np + c4);
            h2 a = {(_Float16)f.x, (_Float16)f.y};
            h2 b = {(_Float16)f.z, (_Float16)f.w};
            *(h2*)(dp + c4)     = a;
            *(h2*)(dp + c4 + 2) = b;
        }
    }
    __syncthreads();

    int m  = lane & 15;
    int kb = (lane & 16) ? 16 : 0;    // A layout K offset for hi lane group
    v8f acc[NT] = {};
    for (int kc = 0; kc < KC_PROJ; ++kc) {
        v16h Bf[NT];
        const _Float16* bbase = projB + ((size_t)kc * NT) * 512 + lane * 16;
#pragma unroll
        for (int nt = 0; nt < NT; ++nt)
            Bf[nt] = *(const v16h*)(bbase + nt * 512);
        A16 A;
        B8 lo, hi;
        lo.v = *(const v8h*)&nsm[wave][m * D_CTX + kc * 32 + kb];
        hi.v = *(const v8h*)&nsm[wave][m * D_CTX + kc * 32 + 8 + kb];
#pragma unroll
        for (int q = 0; q < 4; ++q) { A.p[q] = lo.p[q]; A.p[4 + q] = hi.p[q]; }
#pragma unroll
        for (int nt = 0; nt < NT; ++nt)
            acc[nt] = __builtin_amdgcn_wmma_f32_16x16x32_f16(
                false, A.v, false, Bf[nt], (short)0, acc[nt], false, false);
    }
    // store C: lane<16 -> M=r, lane>=16 -> M=8+r ; N = lane&15
    int mo = (lane & 16) ? 8 : 0;
#pragma unroll
    for (int nt = 0; nt < NT; ++nt) {
        int col = nt * 16 + (lane & 15);
#pragma unroll
        for (int r = 0; r < 8; ++r)
            proj[(rowBase + mo + r) * PROJ_STRIDE + col] = acc[nt][r];
    }
}

// ============================================================================
// zero-init rel_f32 (incl. padding cols), smax (encoded-max identity), denom
// ============================================================================
__global__ void init_kernel(float* __restrict__ rel32,
                            unsigned* __restrict__ smax,
                            float* __restrict__ denom) {
    size_t idx = (size_t)blockIdx.x * blockDim.x + threadIdx.x;
    if (idx < (size_t)N_NODES * REL_STRIDE) rel32[idx] = 0.0f;
    if (idx < 2ull * N_NODES) { smax[idx] = 0u; denom[idx] = 0.0f; }
}

// per-edge score + segment max (monotone float->uint encoding + atomicMax)
__global__ void score_kernel(const float* __restrict__ proj,
                             const float* __restrict__ eInter,
                             const float* __restrict__ eIntra,
                             const int* __restrict__ dstInter,
                             const int* __restrict__ dstIntra,
                             unsigned* __restrict__ smax,
                             float* __restrict__ sArr) {
    int e = blockIdx.x * blockDim.x + threadIdx.x;
    int et = blockIdx.y;
    if (e >= N_EDGES) return;
    const float* ef = (et ? eIntra : eInter) + (size_t)e * D_REL;
    int d = (et ? dstIntra : dstInter)[e];
    const float* pr = proj + (size_t)d * PROJ_STRIDE;
    float s = 0.0f;
    for (int f = 0; f < D_REL; ++f) s += pr[f] * ef[f];
    sArr[(size_t)et * N_EDGES + e] = s;
    unsigned u = __float_as_uint(s);
    u = (u & 0x80000000u) ? ~u : (u | 0x80000000u);
    atomicMax(&smax[(size_t)et * N_NODES + d], u);
}

// exp(s - smax[dst]) + segment sum
__global__ void expsum_kernel(const int* __restrict__ dstInter,
                              const int* __restrict__ dstIntra,
                              const unsigned* __restrict__ smax,
                              const float* __restrict__ sArr,
                              float* __restrict__ exArr,
                              float* __restrict__ denom) {
    int e = blockIdx.x * blockDim.x + threadIdx.x;
    int et = blockIdx.y;
    if (e >= N_EDGES) return;
    int d = (et ? dstIntra : dstInter)[e];
    unsigned u = smax[(size_t)et * N_NODES + d];
    float mx = (u & 0x80000000u) ? __uint_as_float(u & 0x7FFFFFFFu)
                                 : __uint_as_float(~u);
    float ex = __expf(sArr[(size_t)et * N_EDGES + e] - mx);
    exArr[(size_t)et * N_EDGES + e] = ex;
    atomicAdd(&denom[(size_t)et * N_NODES + d], ex);
}

// attention-weighted scatter-add into rel[N,194] (inter->cols 0..96, intra->97..193)
__global__ void accum_kernel(const float* __restrict__ eInter,
                             const float* __restrict__ eIntra,
                             const int* __restrict__ dstInter,
                             const int* __restrict__ dstIntra,
                             const float* __restrict__ exArr,
                             const float* __restrict__ denom,
                             float* __restrict__ rel32) {
    int e = blockIdx.x * blockDim.x + threadIdx.x;
    int et = blockIdx.y;
    if (e >= N_EDGES) return;
    const float* ef = (et ? eIntra : eInter) + (size_t)e * D_REL;
    int d = (et ? dstIntra : dstInter)[e];
    float alpha = exArr[(size_t)et * N_EDGES + e] /
                  denom[(size_t)et * N_NODES + d];
    float* row = rel32 + (size_t)d * REL_STRIDE + et * D_REL;
    for (int f = 0; f < D_REL; ++f)
        atomicAdd(&row[f], ef[f] * alpha);
}

__global__ void cvt_kernel(const float* __restrict__ rel32,
                           _Float16* __restrict__ rel16) {
    size_t idx = (size_t)blockIdx.x * blockDim.x + threadIdx.x;
    if (idx < (size_t)N_NODES * REL_STRIDE) rel16[idx] = (_Float16)rel32[idx];
}

// ============================================================================
// edge_gate: implicit GEMM  gate = P @ Wt  (K = 194*194, N=97 pad 112),
// P[e,(i,j)] = s_rel[e,i]*d_rel[e,j] built on-the-fly from LDS.
// 4 waves/block, 16 edges/wave. s_rel/d_rel rows are gathered into LDS with
// GLOBAL_LOAD_ASYNC_TO_LDS_B128 (ASYNCcnt path, no VGPR transit).
// Fused bias + sigmoid + (e + sig*e) epilogue.
// ============================================================================
__global__ void __launch_bounds__(128)
gate_kernel(const _Float16* __restrict__ rel16,
            const _Float16* __restrict__ Wt,
            const float* __restrict__ eInter,
            const float* __restrict__ eIntra,
            const int* __restrict__ srcInter,
            const int* __restrict__ srcIntra,
            const int* __restrict__ dstInter,
            const int* __restrict__ dstIntra,
            const float* __restrict__ bias,
            float* __restrict__ out) {
    __shared__ __align__(16) _Float16 smem[2][4][16 * SD_STRIDE];
    int wave = threadIdx.x >> 5, lane = threadIdx.x & 31;
    int et = blockIdx.y;
    int edgeBase = blockIdx.x * 64 + wave * 16;
    const int* srcp = et ? srcIntra : srcInter;
    const int* dstp = et ? dstIntra : dstInter;
    const float* efeat = et ? eIntra : eInter;

    // per-lane edge indices for this wave's 16-edge tile, broadcast via shuffle
    int rowIdxS = srcp[edgeBase + (lane & 15)];
    int rowIdxD = dstp[edgeBase + (lane & 15)];

    // stage s_rel / d_rel rows: 26 x 16B async gathers per row + 3 zero chunks
    for (int t = lane; t < 16 * 29; t += 32) {
        int row = t / 29, c = t % 29;
        unsigned ldsS = (unsigned)(uintptr_t)&smem[0][wave][row * SD_STRIDE + c * 8];
        unsigned ldsD = (unsigned)(uintptr_t)&smem[1][wave][row * SD_STRIDE + c * 8];
        if (c < 26) {
            int sNode = __shfl(rowIdxS, row, 32);
            int dNode = __shfl(rowIdxD, row, 32);
            unsigned long long gS = (unsigned long long)(uintptr_t)
                (rel16 + (size_t)sNode * REL_STRIDE + c * 8);
            unsigned long long gD = (unsigned long long)(uintptr_t)
                (rel16 + (size_t)dNode * REL_STRIDE + c * 8);
            asm volatile("global_load_async_to_lds_b128 %0, %1, off"
                         :: "v"(ldsS), "v"(gS) : "memory");
            asm volatile("global_load_async_to_lds_b128 %0, %1, off"
                         :: "v"(ldsD), "v"(gD) : "memory");
        } else {
            int4 z = make_int4(0, 0, 0, 0);
            *(int4*)&smem[0][wave][row * SD_STRIDE + c * 8] = z;
            *(int4*)&smem[1][wave][row * SD_STRIDE + c * 8] = z;
        }
    }
    asm volatile("s_wait_asynccnt 0x0" ::: "memory");
    __syncthreads();

    const _Float16* sS = &smem[0][wave][0];
    const _Float16* sD = &smem[1][wave][0];
    int m  = lane & 15;
    int kb = (lane & 16) ? 16 : 0;
    v8f acc[NT] = {};

    for (int i = 0; i < D_PAIR; ++i) {
        _Float16 sv = sS[m * SD_STRIDE + i];
        h2 s2 = {sv, sv};
#pragma unroll 2
        for (int jc = 0; jc < JC_GATE; ++jc) {
            int kstep = i * JC_GATE + jc;
            const _Float16* bbase = Wt + (size_t)kstep * (NT * 512) + lane * 16;
            v16h Bf[NT];
#pragma unroll
            for (int nt = 0; nt < NT; ++nt)
                Bf[nt] = *(const v16h*)(bbase + nt * 512);
            if (kstep + 1 < KSTEPS_GATE)
                __builtin_prefetch(bbase + NT * 512, 0, 0);
            // on-the-fly rank-1 A fragment: A[m][kk] = s[m,i] * d[m, jc*32+kk]
            B8 lo, hi;
            lo.v = *(const v8h*)&sD[m * SD_STRIDE + jc * 32 + kb];
            hi.v = *(const v8h*)&sD[m * SD_STRIDE + jc * 32 + 8 + kb];
            A16 A;
#pragma unroll
            for (int q = 0; q < 4; ++q) {
                A.p[q]     = s2 * lo.p[q];
                A.p[4 + q] = s2 * hi.p[q];
            }
#pragma unroll
            for (int nt = 0; nt < NT; ++nt)
                acc[nt] = __builtin_amdgcn_wmma_f32_16x16x32_f16(
                    false, A.v, false, Bf[nt], (short)0, acc[nt], false, false);
        }
    }

    // epilogue: out = e + sigmoid(gate + bias) * e
    float* outB = out + (size_t)et * N_EDGES * D_REL;
    int mo = (lane & 16) ? 8 : 0;
#pragma unroll
    for (int nt = 0; nt < NT; ++nt) {
        int col = nt * 16 + (lane & 15);
        if (col < D_REL) {
            float bv = bias[col];
#pragma unroll
            for (int r = 0; r < 8; ++r) {
                int edge = edgeBase + mo + r;
                float g = acc[nt][r] + bv;
                float ev = efeat[(size_t)edge * D_REL + col];
                float sg = 1.0f / (1.0f + __expf(-g));
                outB[(size_t)edge * D_REL + col] = ev + sg * ev;
            }
        }
    }
}

// ============================================================================
extern "C" void kernel_launch(void* const* d_in, const int* in_sizes, int n_in,
                              void* d_out, int out_size, void* d_ws, size_t ws_size,
                              hipStream_t stream) {
    const float* nmat     = (const float*)d_in[0];
    const float* eInter   = (const float*)d_in[1];
    const float* eIntra   = (const float*)d_in[2];
    const float* Wbl      = (const float*)d_in[3];   // [1,768,97]
    const float* Wbl1     = (const float*)d_in[4];   // [97,194,194]
    const float* bbl1     = (const float*)d_in[5];   // [97]
    const int*   srcInter = (const int*)d_in[6];
    const int*   dstInter = (const int*)d_in[7];
    const int*   srcIntra = (const int*)d_in[8];
    const int*   dstIntra = (const int*)d_in[9];
    float* out = (float*)d_out;
    char*  ws  = (char*)d_ws;

    // workspace layout (all offsets are 256B-aligned)
    size_t oProjB = 0;                                            // 24*7*1024 B
    size_t oWt    = oProjB + (size_t)KC_PROJ * NT * 1024;         // 1358*7*1024 B
    size_t oProj  = oWt + (size_t)KSTEPS_GATE * NT * 1024;
    size_t oRel32 = oProj + (size_t)N_NODES * PROJ_STRIDE * 4;
    size_t oRel16 = oRel32 + (size_t)N_NODES * REL_STRIDE * 4;
    size_t oSmax  = oRel16 + (size_t)N_NODES * REL_STRIDE * 2;
    size_t oDenom = oSmax + 2ull * N_NODES * 4;
    size_t oS     = oDenom + 2ull * N_NODES * 4;
    size_t oEx    = oS + 2ull * N_EDGES * 4;

    _Float16* projB = (_Float16*)(ws + oProjB);
    _Float16* Wt    = (_Float16*)(ws + oWt);
    float*    proj  = (float*)(ws + oProj);
    float*    rel32 = (float*)(ws + oRel32);
    _Float16* rel16 = (_Float16*)(ws + oRel16);
    unsigned* smax  = (unsigned*)(ws + oSmax);
    float*    denom = (float*)(ws + oDenom);
    float*    sArr  = (float*)(ws + oS);
    float*    exArr = (float*)(ws + oEx);

    // 1) one-time weight repacks into WMMA-fragment-native f16
    {
        int tot = KC_PROJ * NT * 32;
        pack_projB_kernel<<<(tot + 255) / 256, 256, 0, stream>>>(Wbl, projB);
    }
    {
        int tot = KSTEPS_GATE * NT * 32;
        pack_Wt_kernel<<<(tot + 255) / 256, 256, 0, stream>>>(Wbl1, Wt);
    }

    // 2) proj = n @ W_bl[0]   (WMMA GEMM)
    proj_kernel<<<N_NODES / 32, 64, 0, stream>>>(nmat, projB, proj);

    // 3) node_reduce for both etypes (segment softmax + weighted scatter)
    {
        size_t tot = (size_t)N_NODES * REL_STRIDE;
        init_kernel<<<(unsigned)((tot + 255) / 256), 256, 0, stream>>>(rel32, smax, denom);
    }
    dim3 egrid(N_EDGES / 256, 2);
    score_kernel<<<egrid, 256, 0, stream>>>(proj, eInter, eIntra,
                                            dstInter, dstIntra, smax, sArr);
    expsum_kernel<<<egrid, 256, 0, stream>>>(dstInter, dstIntra, smax,
                                             sArr, exArr, denom);
    accum_kernel<<<egrid, 256, 0, stream>>>(eInter, eIntra, dstInter, dstIntra,
                                            exArr, denom, rel32);
    {
        size_t tot = (size_t)N_NODES * REL_STRIDE;
        cvt_kernel<<<(unsigned)((tot + 255) / 256), 256, 0, stream>>>(rel32, rel16);
    }

    // 4) edge_gate bilinear (implicit GEMM, dominant 478 GFLOP) + epilogue
    dim3 ggrid(N_EDGES / 64, 2);
    gate_kernel<<<ggrid, 128, 0, stream>>>(rel16, Wt, eInter, eIntra,
                                           srcInter, srcIntra, dstInter, dstIntra,
                                           bbl1, out);
}
